// SingleHeadAttention_86225763435068
// MI455X (gfx1250) — matrix operand order repaired and checked by reading, not compiled, not run on hardware
//
#include <hip/hip_runtime.h>

#define B_DIM 4
#define S_DIM 4096
#define E_DIM 1024
#define A_DIM 64

typedef __attribute__((ext_vector_type(16))) __bf16 v16bf;
typedef __attribute__((ext_vector_type(8)))  float  v8f;

union FragBF {
    v16bf  v;
    __bf16 h[16];
    uint4  q[2];
};

__device__ __forceinline__ v8f wmma_bf16(const FragBF& a, const FragBF& b, v8f c) {
    // D = A(16x32 bf16) * B(32x16 bf16) + C(16x16 f32)
    return __builtin_amdgcn_wmma_f32_16x16x32_bf16(
        false, a.v, false, b.v, (short)0, c, false, false);
}

// ---- fragment loaders -------------------------------------------------------
// A operand: row-major tile [16 x 32], rows = output M.
// lane (h = lane>=16, r = lane&15) holds row r, cols {8h..8h+7, 16+8h..16+8h+7}.
__device__ __forceinline__ void load_a_bf16(FragBF& f, const __bf16* p) {
    f.q[0] = *reinterpret_cast<const uint4*>(p);
    f.q[1] = *reinterpret_cast<const uint4*>(p + 16);
}
// B operand: row-major tile [16 x 32], rows = output N (B consumed transposed).
// lane holds row (lane&15), cols {8h..8h+7, 16+8h..16+8h+7}.
__device__ __forceinline__ void load_b_bf16(FragBF& f, const __bf16* p) {
    f.q[0] = *reinterpret_cast<const uint4*>(p);
    f.q[1] = *reinterpret_cast<const uint4*>(p + 16);
}

// ---- stage 0: W f32 -> bf16 (once; W is re-read by every wave in stage 1) ---
__global__ __launch_bounds__(256) void sha_wconv_kernel(
    const float* __restrict__ wq, const float* __restrict__ wk,
    const float* __restrict__ wv,
    unsigned short* __restrict__ wqb_u, unsigned short* __restrict__ wkb_u,
    unsigned short* __restrict__ wvb_u)
{
    const int i = blockIdx.x * 256 + threadIdx.x;        // grid covers A*E
    reinterpret_cast<__bf16*>(wqb_u)[i] = (__bf16)wq[i];
    reinterpret_cast<__bf16*>(wkb_u)[i] = (__bf16)wk[i];
    reinterpret_cast<__bf16*>(wvb_u)[i] = (__bf16)wv[i];
}

// ---- stage 1: Q/K/V projections --------------------------------------------
// One wave -> one 16-row tile of B*S. 12 WMMAs per k=32 step over E.
// Per matrix: 4 fragments (8 x b128) issued before 4 back-to-back WMMAs.
__global__ __launch_bounds__(128) void sha_proj_kernel(
    const float* __restrict__ x,                 // [B*S, E] f32
    const unsigned short* __restrict__ wqb_u,    // bf16 [A, E]
    const unsigned short* __restrict__ wkb_u,
    const unsigned short* __restrict__ wvb_u,
    unsigned short* __restrict__ qout_u,         // bf16 [B*S, A], pre-scaled 1/8
    unsigned short* __restrict__ kout_u,         // bf16 [B*S, A]
    unsigned short* __restrict__ vt_u)           // bf16 [B, A, S] (V transposed)
{
    const __bf16* wqb = reinterpret_cast<const __bf16*>(wqb_u);
    const __bf16* wkb = reinterpret_cast<const __bf16*>(wkb_u);
    const __bf16* wvb = reinterpret_cast<const __bf16*>(wvb_u);
    __bf16* qout = reinterpret_cast<__bf16*>(qout_u);
    __bf16* kout = reinterpret_cast<__bf16*>(kout_u);
    __bf16* vt   = reinterpret_cast<__bf16*>(vt_u);

    const int lane = threadIdx.x & 31;
    const int widx = threadIdx.x >> 5;
    const int row  = lane & 15;
    const int half = lane >> 4;
    const int rowbase = (blockIdx.x * 4 + widx) * 16;

    v8f accQ[4], accK[4], accV[4];
#pragma unroll
    for (int t = 0; t < 4; ++t) {
        v8f z = {0.f, 0.f, 0.f, 0.f, 0.f, 0.f, 0.f, 0.f};
        accQ[t] = z; accK[t] = z; accV[t] = z;
    }

    const float* xrow = x + (size_t)(rowbase + row) * E_DIM + 8 * half;
    const size_t wrow = (size_t)row * E_DIM + 8 * half;   // row-part of W offsets

    for (int e0 = 0; e0 < E_DIM; e0 += 32) {
        // issue all 4 X raw loads first
        const float* xp = xrow + e0;
        const float4 t0 = *reinterpret_cast<const float4*>(xp);
        const float4 t1 = *reinterpret_cast<const float4*>(xp + 4);
        const float4 t2 = *reinterpret_cast<const float4*>(xp + 16);
        const float4 t3 = *reinterpret_cast<const float4*>(xp + 20);

        // first W batch in flight before X conversion
        FragBF wf[4];
#pragma unroll
        for (int t = 0; t < 4; ++t)
            load_b_bf16(wf[t], wqb + wrow + (size_t)t * 16 * E_DIM + e0);

        if (e0 + 32 < E_DIM) {
            __builtin_prefetch(xrow + e0 + 32, 0, 3);
            __builtin_prefetch(xrow + e0 + 48, 0, 3);
        }

        // convert X while W loads fly
        FragBF xf;
        {
            const float s[16] = {t0.x,t0.y,t0.z,t0.w, t1.x,t1.y,t1.z,t1.w,
                                 t2.x,t2.y,t2.z,t2.w, t3.x,t3.y,t3.z,t3.w};
#pragma unroll
            for (int i = 0; i < 16; ++i) xf.h[i] = (__bf16)s[i];
        }

#pragma unroll
        for (int t = 0; t < 4; ++t) accQ[t] = wmma_bf16(xf, wf[t], accQ[t]);

#pragma unroll
        for (int t = 0; t < 4; ++t)
            load_b_bf16(wf[t], wkb + wrow + (size_t)t * 16 * E_DIM + e0);
#pragma unroll
        for (int t = 0; t < 4; ++t) accK[t] = wmma_bf16(xf, wf[t], accK[t]);

#pragma unroll
        for (int t = 0; t < 4; ++t)
            load_b_bf16(wf[t], wvb + wrow + (size_t)t * 16 * E_DIM + e0);
#pragma unroll
        for (int t = 0; t < 4; ++t) accV[t] = wmma_bf16(xf, wf[t], accV[t]);
    }

    // C layout: element r -> (M = r + 8*half, N = lane&15)
    const int b     = rowbase >> 12;            // / S_DIM
    const int sbase = rowbase & (S_DIM - 1);
#pragma unroll
    for (int t = 0; t < 4; ++t) {
#pragma unroll
        for (int r = 0; r < 8; ++r) {
            const int M = r + 8 * half;
            const size_t idx = (size_t)(rowbase + M) * A_DIM + t * 16 + row;
            qout[idx] = (__bf16)(accQ[t][r] * 0.125f);  // fold 1/sqrt(A)
            kout[idx] = (__bf16)accK[t][r];
        }
        alignas(16) __bf16 tmp[8];
#pragma unroll
        for (int r = 0; r < 8; ++r) tmp[r] = (__bf16)accV[t][r];
        const size_t vtidx = ((size_t)(b * A_DIM + t * 16 + row)) * S_DIM + sbase + 8 * half;
        *reinterpret_cast<uint4*>(vt + vtidx) = *reinterpret_cast<const uint4*>(tmp);
    }
}

// ---- stage 2: causal flash attention, 64-key blocks -------------------------
// Block = 4 independent waves; wave w owns 16 queries, all 64 features.
// 16 WMMAs per key block (8 QK^T + 8 PV); one softmax pass per 64 keys.
__global__ __launch_bounds__(128) void sha_attn_kernel(
    const unsigned short* __restrict__ qbuf_u,  // bf16 [B*S, A], scaled
    const unsigned short* __restrict__ kbuf_u,  // bf16 [B*S, A]
    const unsigned short* __restrict__ vt_u,    // bf16 [B, A, S]
    float* __restrict__ out)                    // f32 [B, S, A]
{
    const __bf16* qbuf = reinterpret_cast<const __bf16*>(qbuf_u);
    const __bf16* kbuf = reinterpret_cast<const __bf16*>(kbuf_u);
    const __bf16* vt   = reinterpret_cast<const __bf16*>(vt_u);

    const int lane = threadIdx.x & 31;
    const int widx = threadIdx.x >> 5;
    const int row  = lane & 15;
    const int half = lane >> 4;

    const int b     = blockIdx.x / (S_DIM / 64);
    const int qblk  = blockIdx.x % (S_DIM / 64);
    const int qbase = qblk * 64 + widx * 16;             // within batch
    const size_t grow  = (size_t)b * S_DIM + qbase;
    const size_t kbrow = (size_t)b * S_DIM;
    const size_t vbase = (size_t)b * A_DIM * S_DIM;

    __shared__ __bf16 plds[4][16][64];   // per-wave P transpose staging (2KB/wave)

    FragBF qf[2];
#pragma unroll
    for (int hh = 0; hh < 2; ++hh)
        load_a_bf16(qf[hh], qbuf + (grow + row) * A_DIM + hh * 32 + 8 * half);

    v8f O[4];
#pragma unroll
    for (int t = 0; t < 4; ++t) {
        v8f z = {0.f, 0.f, 0.f, 0.f, 0.f, 0.f, 0.f, 0.f};
        O[t] = z;
    }
    const float FINF = __builtin_inff();
    float m[8], l[8];
#pragma unroll
    for (int r = 0; r < 8; ++r) { m[r] = -FINF; l[r] = 0.f; }

    const int nkb = (qbase + 16 + 63) >> 6;   // 64-key causal blocks
    for (int kb = 0; kb < nkb; ++kb) {
        const int k0 = kb * 64;

        // prefetch next key block (K rows + Vt rows) while computing this one
        if (kb + 1 < nkb) {
            __builtin_prefetch(kbuf + (kbrow + k0 + 64 + lane) * A_DIM, 0, 3);
            __builtin_prefetch(kbuf + (kbrow + k0 + 96 + lane) * A_DIM, 0, 3);
            __builtin_prefetch(vt + vbase + (size_t)lane * S_DIM + k0 + 64, 0, 3);
            __builtin_prefetch(vt + vbase + (size_t)(lane + 32) * S_DIM + k0 + 64, 0, 3);
        }

        // issue all 8 K fragments (16 x b128) before the 8 QK^T WMMAs
        FragBF kf[2][4];
#pragma unroll
        for (int hh = 0; hh < 2; ++hh)
#pragma unroll
            for (int j = 0; j < 4; ++j)
                load_b_bf16(kf[hh][j], kbuf + (kbrow + k0 + j * 16 + row) * A_DIM
                                             + hh * 32 + 8 * half);
        v8f s[4];
#pragma unroll
        for (int j = 0; j < 4; ++j) {
            v8f z = {0.f,0.f,0.f,0.f,0.f,0.f,0.f,0.f};
            s[j] = z;
        }
#pragma unroll
        for (int hh = 0; hh < 2; ++hh)
#pragma unroll
            for (int j = 0; j < 4; ++j)
                s[j] = wmma_bf16(qf[hh], kf[hh][j], s[j]);

        // mask (causal + reference's qk==0 -> -inf quirk) + online softmax
#pragma unroll
        for (int r = 0; r < 8; ++r) {
            const int M  = r + 8 * half;
            const int qi = qbase + M;
            float a[4];
#pragma unroll
            for (int j = 0; j < 4; ++j) {
                float v = s[j][r];
                a[j] = ((k0 + j * 16 + row) > qi || v == 0.0f) ? -FINF : v;
            }
            float mx = fmaxf(fmaxf(a[0], a[1]), fmaxf(a[2], a[3]));
            mx = fmaxf(mx, __shfl_xor(mx, 1));
            mx = fmaxf(mx, __shfl_xor(mx, 2));
            mx = fmaxf(mx, __shfl_xor(mx, 4));
            mx = fmaxf(mx, __shfl_xor(mx, 8));   // reduce over N in 16-lane half
            const float mnew  = fmaxf(m[r], mx);
            const float alpha = __expf(m[r] - mnew);
            float p[4], rs = 0.f;
#pragma unroll
            for (int j = 0; j < 4; ++j) { p[j] = __expf(a[j] - mnew); rs += p[j]; }
            rs += __shfl_xor(rs, 1);
            rs += __shfl_xor(rs, 2);
            rs += __shfl_xor(rs, 4);
            rs += __shfl_xor(rs, 8);
            l[r] = l[r] * alpha + rs;
            m[r] = mnew;
#pragma unroll
            for (int t = 0; t < 4; ++t) O[t][r] *= alpha;
#pragma unroll
            for (int j = 0; j < 4; ++j)
                plds[widx][M][j * 16 + row] = (__bf16)p[j];
        }

        // P as A operand (same-wave LDS ops are in-order -> no barrier needed)
        FragBF pa0, pa1;
        pa0.q[0] = *reinterpret_cast<const uint4*>(&plds[widx][row][8 * half]);
        pa0.q[1] = *reinterpret_cast<const uint4*>(&plds[widx][row][16 + 8 * half]);
        pa1.q[0] = *reinterpret_cast<const uint4*>(&plds[widx][row][32 + 8 * half]);
        pa1.q[1] = *reinterpret_cast<const uint4*>(&plds[widx][row][48 + 8 * half]);

        // issue all 8 Vt fragments before the 8 PV WMMAs
        FragBF vf[4][2];
#pragma unroll
        for (int t = 0; t < 4; ++t) {
            const __bf16* vrow = vt + vbase + (size_t)(t * 16 + row) * S_DIM + k0;
            load_b_bf16(vf[t][0], vrow + 8 * half);
            load_b_bf16(vf[t][1], vrow + 32 + 8 * half);
        }
#pragma unroll
        for (int t = 0; t < 4; ++t) {
            O[t] = wmma_bf16(pa0, vf[t][0], O[t]);
            O[t] = wmma_bf16(pa1, vf[t][1], O[t]);
        }
    }

    // epilogue: divide by row sums, store f32
#pragma unroll
    for (int r = 0; r < 8; ++r) {
        const int M = r + 8 * half;
        const float inv = 1.0f / l[r];
#pragma unroll
        for (int t = 0; t < 4; ++t)
            out[(grow + M) * A_DIM + t * 16 + row] = O[t][r] * inv;
    }
}

extern "C" void kernel_launch(void* const* d_in, const int* in_sizes, int n_in,
                              void* d_out, int out_size, void* d_ws, size_t ws_size,
                              hipStream_t stream) {
    (void)in_sizes; (void)n_in; (void)out_size; (void)ws_size;
    const float* x  = (const float*)d_in[0];   // embedded [B,S,E]
    const float* wq = (const float*)d_in[1];   // [A,E]
    const float* wk = (const float*)d_in[2];
    const float* wv = (const float*)d_in[3];
    float* out = (float*)d_out;                // [B,S,A] f32

    const size_t n_bsa = (size_t)B_DIM * S_DIM * A_DIM;   // 1 Mi elements
    const size_t n_ae  = (size_t)A_DIM * E_DIM;           // 64 Ki elements
    unsigned short* qbuf = (unsigned short*)d_ws;         // bf16, 2 MB
    unsigned short* kbuf = qbuf + n_bsa;                  // bf16, 2 MB
    unsigned short* vtb  = kbuf + n_bsa;                  // bf16, 2 MB
    unsigned short* wqb  = vtb + n_bsa;                   // bf16, 128 KB
    unsigned short* wkb  = wqb + n_ae;
    unsigned short* wvb  = wkb + n_ae;

    sha_wconv_kernel<<<(int)(n_ae / 256), 256, 0, stream>>>(
        wq, wk, wv, wqb, wkb, wvb);
    sha_proj_kernel<<<(B_DIM * S_DIM) / 64, 128, 0, stream>>>(
        x, wqb, wkb, wvb, qbuf, kbuf, vtb);
    sha_attn_kernel<<<B_DIM * (S_DIM / 64), 128, 0, stream>>>(
        qbuf, kbuf, vtb, out);
}
